// ContactGuidedAugment_59717225283903
// MI455X (gfx1250) — matrix operand order repaired and checked by reading, not compiled, not run on hardware
//
#include <hip/hip_runtime.h>
#include <hip/hip_bf16.h>

// ---------------------------------------------------------------------------
// ContactGuidedAugment on MI455X (gfx1250, wave32, WMMA bf16 16x16x32)
//
// S=4096, IN=1024, NODE=512, H=8, D=64
//   1) convert X -> bf16, W{q,k,v,o} -> bf16 transposed ([K][N] for B operand)
//   2) QKV GEMM: wave computes 16x64 (4 B tiles share one A fragment)
//   3) fused flash attention: 1 block = 16 query rows x 8 heads (1 wave/head);
//      contact tile streamed into ping-pong LDS buffers via CDNA5
//      GLOBAL_LOAD_ASYNC_TO_LDS_B128 (ASYNCcnt), overlapping the previous
//      tile's WMMAs; P routed C->A layout through per-wave LDS + s_wait_dscnt
//   4) out-proj (O flat [4096][512] == the reference's scrambled reshape) +
//      bias + residual + LayerNorm fused via 32KB LDS row buffer.
// ---------------------------------------------------------------------------

typedef unsigned short u16;
typedef __bf16  bf16;
typedef bf16    v16bf __attribute__((ext_vector_type(16)));
typedef float   v8f   __attribute__((ext_vector_type(8)));
typedef u16     v8us  __attribute__((ext_vector_type(8)));
typedef u16     v16us __attribute__((ext_vector_type(16)));

#define S_LEN 4096
#define IN_DIM 1024
#define NODE 512
#define HEADS 8
#define HDIM 64

static __device__ __forceinline__ u16 f2bf(float f) {
    unsigned x = __builtin_bit_cast(unsigned, f);
    unsigned r = x + 0x7FFFu + ((x >> 16) & 1u);   // round-to-nearest-even
    return (u16)(r >> 16);
}

static __device__ __forceinline__ v8f zero8() {
    v8f z;
#pragma unroll
    for (int i = 0; i < 8; ++i) z[i] = 0.0f;
    return z;
}

// A operand (16x32, bf16): lane holds row m = lane&15; half = lane>>4.
// element e -> k = (e&8)*2 + half*8 + (e&7). Source row-major, 2x16B loads.
static __device__ __forceinline__ v16bf load_a32(const u16* __restrict__ base,
                                                 int ld, int row_base, int k_base,
                                                 int lane) {
    const int m = lane & 15, half = lane >> 4;
    const u16* p = base + (size_t)(row_base + m) * ld + k_base + half * 8;
    v8us lo = *(const v8us*)p;          // k = half*8 + 0..7
    v8us hi = *(const v8us*)(p + 16);   // k = 16 + half*8 + 0..7
    v16us t;
#pragma unroll
    for (int i = 0; i < 8; ++i) { t[i] = lo[i]; t[i + 8] = hi[i]; }
    return __builtin_bit_cast(v16bf, t);
}

// B operand (32x16, bf16): lane holds column n = lane&15; half = lane>>4.
// element e -> k = (e&8)*2 + half*8 + (e&7). Source K-major [K][N], strided gather.
static __device__ __forceinline__ v16bf load_b32(const u16* __restrict__ base,
                                                 int ld, int k_base, int n_base,
                                                 int lane) {
    const int n = lane & 15, half = lane >> 4;
    const u16* p = base + (size_t)(k_base + half * 8) * ld + n_base + n;
    v16us t;
#pragma unroll
    for (int i = 0; i < 8; ++i) {
        t[i] = p[(size_t)i * ld];
        t[i + 8] = p[(size_t)(16 + i) * ld];
    }
    return __builtin_bit_cast(v16bf, t);
}

static __device__ __forceinline__ v8f wmma_bf16(v16bf a, v16bf b, v8f c) {
    // D = A(16x32 bf16) x B(32x16 bf16) + C(16x16 f32)
    return __builtin_amdgcn_wmma_f32_16x16x32_bf16(
        false, a, false, b, (short)0, c, false, false);
}

static __device__ __forceinline__ float red16_max(float v) {
    v = fmaxf(v, __shfl_xor(v, 1, 32));
    v = fmaxf(v, __shfl_xor(v, 2, 32));
    v = fmaxf(v, __shfl_xor(v, 4, 32));
    v = fmaxf(v, __shfl_xor(v, 8, 32));
    return v;
}
static __device__ __forceinline__ float red16_sum(float v) {
    v += __shfl_xor(v, 1, 32);
    v += __shfl_xor(v, 2, 32);
    v += __shfl_xor(v, 4, 32);
    v += __shfl_xor(v, 8, 32);
    return v;
}

// CDNA5 async global->LDS copy, 16B per issuing lane (ASYNCcnt tracked).
// LDS address = low 32 bits of the generic shared pointer (ISA 10.2 aperture).
static __device__ __forceinline__ void async_copy_b128(float* lds_dst,
                                                       const float* gsrc) {
    unsigned lds = (unsigned)(size_t)lds_dst;
    asm volatile("global_load_async_to_lds_b128 %0, %1, off"
                 :: "v"(lds), "v"(gsrc) : "memory");
}
static __device__ __forceinline__ void wait_async0() {
    asm volatile("s_wait_asynccnt 0" ::: "memory");
}

// ------------------------------ converts -----------------------------------

__global__ void __launch_bounds__(256)
k_cvt_bf16(const float* __restrict__ src, u16* __restrict__ dst, int n) {
    int stride = gridDim.x * blockDim.x;
    for (int i = blockIdx.x * blockDim.x + threadIdx.x; i < n; i += stride)
        dst[i] = f2bf(src[i]);
}

// W[rows][cols] f32 -> Wt[cols][rows] bf16  (B operand wants [K][N])
__global__ void __launch_bounds__(256)
k_cvt_transpose(const float* __restrict__ W, u16* __restrict__ Wt,
                int rows, int cols) {
    int total = rows * cols;
    int stride = gridDim.x * blockDim.x;
    for (int i = blockIdx.x * blockDim.x + threadIdx.x; i < total; i += stride) {
        int r = i / cols, c = i % cols;
        Wt[(size_t)c * rows + r] = f2bf(W[i]);
    }
}

// ------------------------------ QKV GEMM -----------------------------------
// grid: (S/16, NODE/256, 3)  block: 128 (4 waves); each wave owns 16x64 of
// the output so one A fragment feeds 4 WMMAs.
__global__ void __launch_bounds__(128)
k_qkv_gemm(const u16* __restrict__ Xbf,                       // [S][IN]
           const u16* __restrict__ Wqt, const u16* __restrict__ Wkt,
           const u16* __restrict__ Wvt,                       // [IN][NODE]
           const float* __restrict__ bq, const float* __restrict__ bk,
           const float* __restrict__ bv,
           u16* __restrict__ Qbf,   // [H][S][D]
           u16* __restrict__ Ktbf,  // [H][D][S]
           u16* __restrict__ Vbf)   // [H][S][D]
{
    const int lane = threadIdx.x & 31;
    const int wave = threadIdx.x >> 5;
    const int row_base = blockIdx.x * 16;
    const int colw = (blockIdx.y * 4 + wave) * 64;   // 64 output cols per wave
    const int z = blockIdx.z;
    const u16* Wt = (z == 0) ? Wqt : (z == 1) ? Wkt : Wvt;
    const float* bias = (z == 0) ? bq : (z == 1) ? bk : bv;

    v8f acc[4];
#pragma unroll
    for (int j = 0; j < 4; ++j) acc[j] = zero8();

#pragma unroll 4
    for (int k = 0; k < IN_DIM; k += 32) {
        v16bf a = load_a32(Xbf, IN_DIM, row_base, k, lane);
#pragma unroll
        for (int j = 0; j < 4; ++j)
            acc[j] = wmma_bf16(a, load_b32(Wt, NODE, k, colw + j * 16, lane),
                               acc[j]);
    }

    const int half = lane >> 4;
#pragma unroll
    for (int j = 0; j < 4; ++j) {
        const int col = colw + j * 16 + (lane & 15);
        const float bb = bias[col];
        const int h = col >> 6;       // head
        const int d = col & 63;       // dim within head
#pragma unroll
        for (int r = 0; r < 8; ++r) {
            const int s = row_base + r + 8 * half;
            const u16 v = f2bf(acc[j][r] + bb);
            if (z == 0)      Qbf [((size_t)h * S_LEN + s) * HDIM + d] = v;
            else if (z == 1) Ktbf[((size_t)h * HDIM + d) * S_LEN + s] = v;
            else             Vbf [((size_t)h * S_LEN + s) * HDIM + d] = v;
        }
    }
}

// --------------------------- fused flash attention --------------------------
// grid: S/16 blocks; block: 256 threads = 8 waves; wave h handles head h for
// the same 16 query rows. Contact tiles (16x32 f32) double-buffered in LDS,
// streamed by GLOBAL_LOAD_ASYNC_TO_LDS_B128 overlapping previous tile's math.
__global__ void __launch_bounds__(256)
k_attn(const u16* __restrict__ Qbf,   // [H][S][D]
       const u16* __restrict__ Ktbf,  // [H][D][S]
       const u16* __restrict__ Vbf,   // [H][S][D]
       const float* __restrict__ contact,  // [S][S]
       const float* __restrict__ Wc,       // [H]
       u16* __restrict__ Obf)              // [H][S][D] bf16 (flat = scrambled)
{
    __shared__ __align__(16) float sContact[2][16 * 32];   // 4 KB ping-pong
    __shared__ __align__(16) u16   sP[HEADS][16 * 32];     // 8 KB

    const int tid  = threadIdx.x;
    const int lane = tid & 31;
    const int h    = tid >> 5;
    const int half = lane >> 4;
    const int nn   = lane & 15;
    const int row_base = blockIdx.x * 16;

    // contact tile async copy: threads 0..127 move 16B each (512 floats)
    const int crow = tid >> 3;          // 0..15
    const int ccol = (tid & 7) * 4;     // 0,4,..,28
    const float* gC = contact + (size_t)(row_base + crow) * S_LEN + ccol;
    const bool copier = (tid < 128);

    const float scale = Wc[h] * 0.125f;          // wc / sqrt(64)
    const u16* Qh  = Qbf  + (size_t)h * S_LEN * HDIM;
    const u16* Kth = Ktbf + (size_t)h * HDIM  * S_LEN;
    const u16* Vh  = Vbf  + (size_t)h * S_LEN * HDIM;

    // Q tile for this wave, resident for the whole loop: 16 x 64
    const v16bf qa0 = load_a32(Qh, HDIM, row_base, 0,  lane);
    const v16bf qa1 = load_a32(Qh, HDIM, row_base, 32, lane);

    v8f o0 = zero8(), o1 = zero8(), o2 = zero8(), o3 = zero8();
    float mrow[8], lrow[8];
#pragma unroll
    for (int r = 0; r < 8; ++r) { mrow[r] = -__builtin_inff(); lrow[r] = 0.0f; }

    // prologue: stream tile 0 into buffer 0
    if (copier) async_copy_b128(&sContact[0][crow * 32 + ccol], gC);
    int cur = 0;

    for (int kb = 0; kb < S_LEN; kb += 32) {
        // wait for own async copies (targets buf[cur]), then make visible
        wait_async0();
        __syncthreads();
        // prefetch next tile into the other buffer while we compute
        if (copier && (kb + 32 < S_LEN))
            async_copy_b128(&sContact[cur ^ 1][crow * 32 + ccol], gC + kb + 32);
        const float* sC = sContact[cur];
        cur ^= 1;

        // ---- scores S = Q K^T  (two 16-key tiles, K-dim 64 = 2x32) ----
        v8f s0 = zero8(), s1 = zero8();
        s0 = wmma_bf16(qa0, load_b32(Kth, S_LEN, 0,  kb,      lane), s0);
        s0 = wmma_bf16(qa1, load_b32(Kth, S_LEN, 32, kb,      lane), s0);
        s1 = wmma_bf16(qa0, load_b32(Kth, S_LEN, 0,  kb + 16, lane), s1);
        s1 = wmma_bf16(qa1, load_b32(Kth, S_LEN, 32, kb + 16, lane), s1);

        // ---- multiplicative contact mask + online softmax ----
        float rm[8];
#pragma unroll
        for (int r = 0; r < 8; ++r) {
            const int rr = (r + 8 * half) * 32;
            float l0 = s0[r] * scale * sC[rr + nn];
            float l1 = s1[r] * scale * sC[rr + 16 + nn];
            s0[r] = l0; s1[r] = l1;
            rm[r] = red16_max(fmaxf(l0, l1));
        }
#pragma unroll
        for (int r = 0; r < 8; ++r) {
            float mn = fmaxf(mrow[r], rm[r]);
            float alpha = __expf(mrow[r] - mn);
            mrow[r] = mn;
            float p0 = __expf(s0[r] - mn);
            float p1 = __expf(s1[r] - mn);
            s0[r] = p0; s1[r] = p1;
            lrow[r] = lrow[r] * alpha + red16_sum(p0 + p1);
            o0[r] *= alpha; o1[r] *= alpha; o2[r] *= alpha; o3[r] *= alpha;
        }

        // ---- C-layout -> A-layout bounce through per-wave LDS ----
        u16* myP = sP[h];
#pragma unroll
        for (int r = 0; r < 8; ++r) {
            const int rr = (r + 8 * half) * 32;
            myP[rr + nn]      = f2bf(s0[r]);
            myP[rr + 16 + nn] = f2bf(s1[r]);
        }
        asm volatile("s_wait_dscnt 0" ::: "memory");  // in-wave DS RAW fence
        const v16bf pa = load_a32(myP, 32, 0, 0, lane);

        // ---- O += P @ V  (K = 32 keys, four 16-dim tiles) ----
        o0 = wmma_bf16(pa, load_b32(Vh, HDIM, kb, 0,  lane), o0);
        o1 = wmma_bf16(pa, load_b32(Vh, HDIM, kb, 16, lane), o1);
        o2 = wmma_bf16(pa, load_b32(Vh, HDIM, kb, 32, lane), o2);
        o3 = wmma_bf16(pa, load_b32(Vh, HDIM, kb, 48, lane), o3);
    }

    // ---- normalize and store O (bf16, flat [H][S][D]) ----
#pragma unroll
    for (int r = 0; r < 8; ++r) {
        const float inv = 1.0f / lrow[r];
        const int s = row_base + r + 8 * half;
        const size_t base = ((size_t)h * S_LEN + s) * HDIM;
        Obf[base + 0  + nn] = f2bf(o0[r] * inv);
        Obf[base + 16 + nn] = f2bf(o1[r] * inv);
        Obf[base + 32 + nn] = f2bf(o2[r] * inv);
        Obf[base + 48 + nn] = f2bf(o3[r] * inv);
    }
}

// -------------------- out-proj + residual + LayerNorm -----------------------
// O flat [H][S][D] reinterpreted as [4096][512] == the reference's scrambled
// reshape, so A rows are simply contiguous 512-element chunks of Obf.
// grid: S/16 blocks; block: 256 (8 waves, each owns 64 output cols).
__global__ void __launch_bounds__(256)
k_proj_ln(const u16* __restrict__ Obf,   // viewed [4096][512]
          const u16* __restrict__ Wot,   // [512][512]  (K-major)
          const float* __restrict__ bo,
          const float* __restrict__ X,   // [S][IN], residual = X[:, :512]
          const float* __restrict__ gamma, const float* __restrict__ beta,
          float* __restrict__ out)       // [S][512]
{
    __shared__ float sRow[16 * NODE];    // 32 KB
    const int lane = threadIdx.x & 31;
    const int wave = threadIdx.x >> 5;
    const int half = lane >> 4;
    const int nn   = lane & 15;
    const int row_base = blockIdx.x * 16;
    const int colw = wave * 64;

    v8f acc[4];
#pragma unroll
    for (int j = 0; j < 4; ++j) acc[j] = zero8();

#pragma unroll 4
    for (int k = 0; k < NODE; k += 32) {
        v16bf a = load_a32(Obf, NODE, row_base, k, lane);
#pragma unroll
        for (int j = 0; j < 4; ++j)
            acc[j] = wmma_bf16(a, load_b32(Wot, NODE, k, colw + j * 16, lane),
                               acc[j]);
    }

    // epilogue: bias + residual into LDS row buffer
#pragma unroll
    for (int j = 0; j < 4; ++j) {
        const int col = colw + j * 16 + nn;
        const float bb = bo[col];
#pragma unroll
        for (int r = 0; r < 8; ++r) {
            const int srow = r + 8 * half;
            float v = acc[j][r] + bb +
                      X[(size_t)(row_base + srow) * IN_DIM + col];
            sRow[srow * NODE + col] = v;
        }
    }
    __syncthreads();

    // LayerNorm: each wave handles 2 rows (16 lanes per row)
    const int srow = wave * 2 + half;
    const float* rowp = &sRow[srow * NODE];
    float sum = 0.0f;
#pragma unroll
    for (int j = 0; j < 32; ++j) sum += rowp[nn + 16 * j];
    const float mean = red16_sum(sum) * (1.0f / (float)NODE);

    float vs = 0.0f;
#pragma unroll
    for (int j = 0; j < 32; ++j) {
        float d = rowp[nn + 16 * j] - mean;
        vs += d * d;
    }
    const float rstd = rsqrtf(red16_sum(vs) * (1.0f / (float)NODE) + 1e-5f);

    const size_t orow = (size_t)(row_base + srow) * NODE;
#pragma unroll
    for (int j = 0; j < 32; ++j) {
        const int col = nn + 16 * j;
        out[orow + col] = (rowp[col] - mean) * rstd * gamma[col] + beta[col];
    }
}

// ------------------------------ launcher ------------------------------------

extern "C" void kernel_launch(void* const* d_in, const int* in_sizes, int n_in,
                              void* d_out, int out_size, void* d_ws, size_t ws_size,
                              hipStream_t stream) {
    const float* X       = (const float*)d_in[0];   // [4096][1024]
    const float* contact = (const float*)d_in[1];   // [4096][4096]
    const float* Wq      = (const float*)d_in[2];   // [512][1024]
    const float* bq      = (const float*)d_in[3];
    const float* Wk      = (const float*)d_in[4];
    const float* bk      = (const float*)d_in[5];
    const float* Wv      = (const float*)d_in[6];
    const float* bv      = (const float*)d_in[7];
    const float* Wc      = (const float*)d_in[8];   // [H]
    const float* Wo      = (const float*)d_in[9];   // [512][512]
    const float* bo      = (const float*)d_in[10];
    const float* gamma   = (const float*)d_in[11];
    const float* beta    = (const float*)d_in[12];
    float* out = (float*)d_out;

    // workspace carve-up (bytes, 256-aligned)
    char* ws = (char*)d_ws;
    size_t off = 0;
    auto alloc = [&](size_t bytes) {
        char* p = ws + off;
        off += (bytes + 255) & ~(size_t)255;
        return p;
    };
    u16* Xbf  = (u16*)alloc((size_t)S_LEN * IN_DIM * 2);        // 8 MB
    u16* Wqt  = (u16*)alloc((size_t)IN_DIM * NODE * 2);         // 1 MB
    u16* Wkt  = (u16*)alloc((size_t)IN_DIM * NODE * 2);         // 1 MB
    u16* Wvt  = (u16*)alloc((size_t)IN_DIM * NODE * 2);         // 1 MB
    u16* Wot  = (u16*)alloc((size_t)NODE * NODE * 2);           // 0.5 MB
    u16* Qbf  = (u16*)alloc((size_t)HEADS * S_LEN * HDIM * 2);  // 4 MB
    u16* Ktbf = (u16*)alloc((size_t)HEADS * HDIM * S_LEN * 2);  // 4 MB
    u16* Vbf  = (u16*)alloc((size_t)HEADS * S_LEN * HDIM * 2);  // 4 MB
    u16* Obf  = (u16*)alloc((size_t)HEADS * S_LEN * HDIM * 2);  // 4 MB
    (void)ws_size; (void)in_sizes; (void)n_in; (void)out_size;

    // 1) precision conversion / transposes
    k_cvt_bf16<<<1024, 256, 0, stream>>>(X, Xbf, S_LEN * IN_DIM);
    k_cvt_transpose<<<1024, 256, 0, stream>>>(Wq, Wqt, NODE, IN_DIM);
    k_cvt_transpose<<<1024, 256, 0, stream>>>(Wk, Wkt, NODE, IN_DIM);
    k_cvt_transpose<<<1024, 256, 0, stream>>>(Wv, Wvt, NODE, IN_DIM);
    k_cvt_transpose<<<512, 256, 0, stream>>>(Wo, Wot, NODE, NODE);

    // 2) QKV projections (wave = 16x64 tile, 4x A reuse)
    k_qkv_gemm<<<dim3(S_LEN / 16, NODE / 256, 3), 128, 0, stream>>>(
        Xbf, Wqt, Wkt, Wvt, bq, bk, bv, Qbf, Ktbf, Vbf);

    // 3) fused contact-masked flash attention (async LDS contact streaming)
    k_attn<<<S_LEN / 16, 256, 0, stream>>>(Qbf, Ktbf, Vbf, contact, Wc, Obf);

    // 4) out-proj + residual + LayerNorm
    k_proj_ln<<<S_LEN / 16, 256, 0, stream>>>(Obf, Wot, bo, X, gamma, beta, out);
}
